// HeteroRGCN_69870527971338
// MI455X (gfx1250) — compile-verified, matrix operand order checked by reading, not run on hardware
//
#include <hip/hip_runtime.h>

// ---------------------------------------------------------------------------
// HeteroRGCN forward for MI455X (gfx1250), wave32 + WMMA bf16.
//
// Per layer, ONE fused GEMM:
//     hcat[N x 576] = feat[N x K](bf16) @ Wcat[K x 576](bf16)
// with Wcat = [W_0 | ... | W_7 | root], computed via v_wmma_f32_16x16x32_bf16.
// A-fragments are loaded DIRECTLY from global (row-major A matches the
// documented 16-bit A VGPR layout: two contiguous 16B runs per lane), the
// B panel (K x 64, <=16KB) is staged to LDS once per block -> the K-loop has
// no barriers. Partial tiles are handled by index clamping (stores stay
// guarded), so staging/compute is fully branchless and EXEC stays all-ones
// around every WMMA.
//
// Aggregation: mean-per-relation via precomputed in-degree counts
// cnt[node][rel] (graph is static across layers -> counted once), then a
// single atomic edge pass per layer scaled by 1/cnt. relu+BN fused into the
// bf16 re-cast feeding the next GEMM.
//
// Workspace (~143 MB): hcat N*576 f32 | acc N*64 f32 | hb N*128 bf16 |
//                      wb 128*576 bf16 | wlb 64*16 bf16 | cnt N*8 i32
// ---------------------------------------------------------------------------

typedef __attribute__((ext_vector_type(16))) __bf16 v16bf;
typedef __attribute__((ext_vector_type(8)))  __bf16 v8bf;
typedef __attribute__((ext_vector_type(8)))  float  v8f;

#define NNODES 50000
#define NEDGES 800000
#define FIN    128
#define HDIM   64
#define NCLS   16
#define NREL   8
#define NCAT   ((NREL + 1) * HDIM)   // 576
#define BN_EPS 1e-5f

// GEMM tiling: 256 threads = 8 waves; each wave owns a 16x64 output strip.
#define MT   128
#define NT   64
#define KMAX 128
#define BPAD 8      // LDS row padding (bf16) -> 272B stride, conflict-free b128

// ---------------------------------------------------------------------------
// WMMA GEMM: C[M x NC] = A[M x K] @ B[K x NC] (+ bias per column, optional)
// A, B bf16 row-major; C f32 row-major. K % 32 == 0, K <= 128. M, NC guarded.
// ---------------------------------------------------------------------------
__global__ __launch_bounds__(256) void gemm_bf16_wmma(
    const __bf16* __restrict__ A, const __bf16* __restrict__ B,
    const float* __restrict__ bias, float* __restrict__ C,
    int M, int K, int NC)
{
  __shared__ __bf16 Bs[NT][KMAX + BPAD];   // 17.4 KB

  const int tid  = threadIdx.x;
  const int wave = tid >> 5;
  const int lane = tid & 31;
  const int hs   = lane >> 4;     // K half-stripe select
  const int l    = lane & 15;
  const int m0   = blockIdx.x * MT;
  const int n0   = blockIdx.y * NT;

  // ---- Stage full B panel [K x NT] transposed into LDS (once, branchless).
  // 16B global chunks of 8 consecutive columns; OOB chunks clamped to col 0
  // (padding columns feed only guarded-out outputs).
  for (int i = tid; i < K * (NT / 8); i += 256) {
    int k  = i >> 3;
    int cg = i & 7;
    int cbase = n0 + cg * 8;
    if (cbase + 8 > NC) cbase = 0;
    v8bf v = *(const v8bf*)&B[(long)k * NC + cbase];
#pragma unroll
    for (int t = 0; t < 8; ++t) Bs[cg * 8 + t][k] = v[t];
  }
  __syncthreads();

  // Row clamp: garbage rows are computed but never stored.
  const int aRow = min(m0 + wave * 16 + l, M - 1);
  const __bf16* Abase = A + (long)aRow * K;

  v8f acc[4] = {v8f{}, v8f{}, v8f{}, v8f{}};

  for (int k0 = 0; k0 < K; k0 += 32) {
    // A fragment straight from global: 16-bit A 16x32 layout = two
    // contiguous 16B runs per lane (lane<16: K{0..7,16..23}; lane>=16:
    // K{8..15,24..31}).  -> 2x global_load_b128, no LDS, no barrier.
    v8bf alo = *(const v8bf*)(Abase + k0 + hs * 8);
    v8bf ahi = *(const v8bf*)(Abase + k0 + 16 + hs * 8);
    v16bf a = __builtin_shufflevector(alo, ahi,
        0, 1, 2, 3, 4, 5, 6, 7, 8, 9, 10, 11, 12, 13, 14, 15);

#pragma unroll
    for (int j = 0; j < 4; ++j) {
      const __bf16* brow = &Bs[j * 16 + l][0];
      v8bf blo = *(const v8bf*)(brow + k0 + hs * 8);
      v8bf bhi = *(const v8bf*)(brow + k0 + 16 + hs * 8);
      v16bf b = __builtin_shufflevector(blo, bhi,
          0, 1, 2, 3, 4, 5, 6, 7, 8, 9, 10, 11, 12, 13, 14, 15);
      acc[j] = __builtin_amdgcn_wmma_f32_16x16x32_bf16(
          false, a, false, b, (short)0, acc[j], false, false);
    }
  }

  // Store: f32 C/D layout -> VGPR r holds row (r + 8*halfsel), col = lane&15.
  const bool fullM = (m0 + MT) <= M;
#pragma unroll
  for (int j = 0; j < 4; ++j) {
    int col = n0 + j * 16 + l;
    if (col < NC) {
      float bv = bias ? bias[col] : 0.0f;
      int rbase = m0 + wave * 16 + hs * 8;
      if (fullM) {
#pragma unroll
        for (int r = 0; r < 8; ++r)
          C[(long)(rbase + r) * NC + col] = acc[j][r] + bv;
      } else {
#pragma unroll
        for (int r = 0; r < 8; ++r)
          if (rbase + r < M) C[(long)(rbase + r) * NC + col] = acc[j][r] + bv;
      }
    }
  }
}

// ---------------------------------------------------------------------------
// Weight packing: Wcat[k][r*64+h] = W[r][k][h]; Wcat[k][512+h] = root[k][h]
// ---------------------------------------------------------------------------
__global__ void pack_wcat_bf16(const float* __restrict__ W,
                               const float* __restrict__ root,
                               __bf16* __restrict__ out, int K)
{
  int i = blockIdx.x * blockDim.x + threadIdx.x;
  int tot = K * NCAT;
  if (i >= tot) return;
  int k = i / NCAT, col = i - k * NCAT;
  float v;
  if (col < NREL * HDIM) {
    int r = col >> 6, h = col & 63;
    v = W[((long)r * K + k) * HDIM + h];
  } else {
    v = root[(long)k * HDIM + (col - NREL * HDIM)];
  }
  out[i] = (__bf16)v;
}

__global__ void cast_f32_bf16(const float* __restrict__ in,
                              __bf16* __restrict__ out, long n)
{
  long i = (long)blockIdx.x * blockDim.x + threadIdx.x;
  long stride = (long)gridDim.x * blockDim.x;
  for (; i < n; i += stride) out[i] = (__bf16)in[i];
}

// ---------------------------------------------------------------------------
// Edge machinery: per-(dst,rel) counts (once), then scaled atomic scatter.
// ---------------------------------------------------------------------------
__global__ void zero_i32(int* __restrict__ p, int n)
{
  int i = blockIdx.x * blockDim.x + threadIdx.x;
  if (i < n) p[i] = 0;
}

__global__ void count_edges(const int* __restrict__ dst,
                            const int* __restrict__ et,
                            int* __restrict__ cnt)
{
  int e = blockIdx.x * blockDim.x + threadIdx.x;
  if (e < NEDGES) atomicAdd(&cnt[dst[e] * NREL + et[e]], 1);
}

// acc[i][h] = root-part + bias  (root block = cols 512..575 of hcat)
__global__ void init_acc(const float* __restrict__ hcat,
                         const float* __restrict__ bias,
                         float* __restrict__ acc)
{
  long idx = (long)blockIdx.x * blockDim.x + threadIdx.x;
  if (idx >= (long)NNODES * HDIM) return;
  int h = idx & 63;
  long i = idx >> 6;
  acc[idx] = hcat[i * NCAT + NREL * HDIM + h] + bias[h];
}

// One pass over all edges; each edge contributes h_r[src] / cnt[dst][r].
// 16 threads per edge, float4 per thread (16B-aligned strides).
__global__ void edge_scatter(const float* __restrict__ hcat,
                             const int* __restrict__ src,
                             const int* __restrict__ dst,
                             const int* __restrict__ et,
                             const int* __restrict__ cnt,
                             float* __restrict__ acc)
{
  long idx = (long)blockIdx.x * blockDim.x + threadIdx.x;
  const long tot = (long)NEDGES * 16;
  const long stride = (long)gridDim.x * blockDim.x;
  for (; idx < tot; idx += stride) {
    int e = (int)(idx >> 4);
    int q = (int)(idx & 15);
    int s = src[e], d = dst[e], r = et[e];
    float w = 1.0f / fmaxf((float)cnt[d * NREL + r], 1.0f);
    const float4 m = *(const float4*)&hcat[(long)s * NCAT + r * HDIM + q * 4];
    float* o = &acc[(long)d * HDIM + q * 4];
    atomicAdd(o + 0, m.x * w);
    atomicAdd(o + 1, m.y * w);
    atomicAdd(o + 2, m.z * w);
    atomicAdd(o + 3, m.w * w);
  }
}

// h = BN(relu(acc)) -> bf16 feature buffer for the next GEMM
__global__ void relu_bn_cast(const float* __restrict__ acc,
                             const float* __restrict__ g,
                             const float* __restrict__ be,
                             const float* __restrict__ rm,
                             const float* __restrict__ rv,
                             __bf16* __restrict__ hb)
{
  long idx = (long)blockIdx.x * blockDim.x + threadIdx.x;
  if (idx >= (long)NNODES * HDIM) return;
  int h = idx & 63;
  float v = fmaxf(acc[idx], 0.0f);
  v = (v - rm[h]) * rsqrtf(rv[h] + BN_EPS) * g[h] + be[h];
  hb[idx] = (__bf16)v;
}

// ---------------------------------------------------------------------------
extern "C" void kernel_launch(void* const* d_in, const int* in_sizes, int n_in,
                              void* d_out, int out_size, void* d_ws, size_t ws_size,
                              hipStream_t stream)
{
  const float* x     = (const float*)d_in[0];
  const int*   ei    = (const int*)  d_in[1];
  const int*   et    = (const int*)  d_in[2];
  const float* W1    = (const float*)d_in[3];
  const float* root1 = (const float*)d_in[4];
  const float* b1    = (const float*)d_in[5];
  const float* W2    = (const float*)d_in[6];
  const float* root2 = (const float*)d_in[7];
  const float* b2    = (const float*)d_in[8];
  const float* W3    = (const float*)d_in[9];
  const float* root3 = (const float*)d_in[10];
  const float* b3    = (const float*)d_in[11];
  const float* g1 = (const float*)d_in[12], *be1 = (const float*)d_in[13];
  const float* rm1= (const float*)d_in[14], *rv1 = (const float*)d_in[15];
  const float* g2 = (const float*)d_in[16], *be2 = (const float*)d_in[17];
  const float* rm2= (const float*)d_in[18], *rv2 = (const float*)d_in[19];
  const float* g3 = (const float*)d_in[20], *be3 = (const float*)d_in[21];
  const float* rm3= (const float*)d_in[22], *rv3 = (const float*)d_in[23];
  const float* Wl = (const float*)d_in[24];
  const float* bl = (const float*)d_in[25];

  const int* src = ei;
  const int* dst = ei + NEDGES;

  // workspace carve-out (256B aligned)
  char* ws = (char*)d_ws;
  size_t off = 0;
  auto carve = [&](size_t bytes) {
    void* p = ws + off;
    off = (off + bytes + 255) & ~(size_t)255;
    return p;
  };
  float*  hcat = (float*) carve((size_t)NNODES * NCAT * 4);
  float*  acc  = (float*) carve((size_t)NNODES * HDIM * 4);
  __bf16* hb   = (__bf16*)carve((size_t)NNODES * FIN * 2);
  __bf16* wb   = (__bf16*)carve((size_t)FIN * NCAT * 2);
  __bf16* wlb  = (__bf16*)carve((size_t)HDIM * NCLS * 2);
  int*    cnt  = (int*)   carve((size_t)NNODES * NREL * 4);
  (void)ws_size; (void)in_sizes; (void)n_in; (void)out_size;

  const int NB = 256;
  const long nodeElems = (long)NNODES * HDIM;
  const int  nodeGrid  = (int)((nodeElems + NB - 1) / NB);
  const int  scatGrid  = 8192;  // grid-stride over E*16
  dim3 ggrid((NNODES + MT - 1) / MT, (NCAT + NT - 1) / NT);

  // --- counts: graph identical across layers, compute once ---
  zero_i32<<<(NNODES * NREL + NB - 1) / NB, NB, 0, stream>>>(cnt, NNODES * NREL);
  count_edges<<<(NEDGES + NB - 1) / NB, NB, 0, stream>>>(dst, et, cnt);

  // --- layer 1 ---
  cast_f32_bf16<<<4096, NB, 0, stream>>>(x, hb, (long)NNODES * FIN);
  pack_wcat_bf16<<<(FIN * NCAT + NB - 1) / NB, NB, 0, stream>>>(W1, root1, wb, FIN);
  gemm_bf16_wmma<<<ggrid, NB, 0, stream>>>(hb, wb, nullptr, hcat, NNODES, FIN, NCAT);
  init_acc<<<nodeGrid, NB, 0, stream>>>(hcat, b1, acc);
  edge_scatter<<<scatGrid, NB, 0, stream>>>(hcat, src, dst, et, cnt, acc);
  relu_bn_cast<<<nodeGrid, NB, 0, stream>>>(acc, g1, be1, rm1, rv1, hb);

  // --- layer 2 ---
  pack_wcat_bf16<<<(HDIM * NCAT + NB - 1) / NB, NB, 0, stream>>>(W2, root2, wb, HDIM);
  gemm_bf16_wmma<<<ggrid, NB, 0, stream>>>(hb, wb, nullptr, hcat, NNODES, HDIM, NCAT);
  init_acc<<<nodeGrid, NB, 0, stream>>>(hcat, b2, acc);
  edge_scatter<<<scatGrid, NB, 0, stream>>>(hcat, src, dst, et, cnt, acc);
  relu_bn_cast<<<nodeGrid, NB, 0, stream>>>(acc, g2, be2, rm2, rv2, hb);

  // --- layer 3 ---
  pack_wcat_bf16<<<(HDIM * NCAT + NB - 1) / NB, NB, 0, stream>>>(W3, root3, wb, HDIM);
  gemm_bf16_wmma<<<ggrid, NB, 0, stream>>>(hb, wb, nullptr, hcat, NNODES, HDIM, NCAT);
  init_acc<<<nodeGrid, NB, 0, stream>>>(hcat, b3, acc);
  edge_scatter<<<scatGrid, NB, 0, stream>>>(hcat, src, dst, et, cnt, acc);
  relu_bn_cast<<<nodeGrid, NB, 0, stream>>>(acc, g3, be3, rm3, rv3, hb);

  // --- final linear: out[N x 16] = h3 @ Wl + bl (bias fused in GEMM) ---
  cast_f32_bf16<<<(HDIM * NCLS + NB - 1) / NB, NB, 0, stream>>>(Wl, wlb, (long)HDIM * NCLS);
  dim3 lgrid((NNODES + MT - 1) / MT, 1);
  gemm_bf16_wmma<<<lgrid, NB, 0, stream>>>(hb, wlb, bl, (float*)d_out,
                                           NNODES, HDIM, NCLS);
}
